// AggBondModule_49572512530563
// MI455X (gfx1250) — matrix operand order repaired and checked by reading, not compiled, not run on hardware
//
#include <hip/hip_runtime.h>

// ---------------------------------------------------------------------------
// Edge-MLP: out[e,:] = relu(concat(node[src[e]], node[dst[e]], edge[e]) @ W + b)
// Irregular-A GEMM [E x 384] @ [384 x 128] on the bf16 WMMA path
// (v_wmma_f32_16x16x32_bf16, fp32 accumulate). W staged to LDS as bf16 in
// B-fragment order; A gathered per-lane from global and converted to bf16.
// Round 2: defeat LICM on the B-fragment ds_loads (they are blk-loop
// invariant) with an asm laundering barrier on the K offset — round 1 showed
// the compiler hoisting all 96 fragments and spilling them to scratch.
// ---------------------------------------------------------------------------

typedef __bf16 v16bf __attribute__((ext_vector_type(16)));
typedef float  v8f   __attribute__((ext_vector_type(8)));

#define D_NODE 128
#define KTOT   384          // 2*D_NODE + D_EDGE
#define TPB    256          // 8 waves per workgroup
#define WAVES  8

__global__ __launch_bounds__(TPB) void agg_bond_wmma(
    const float* __restrict__ node_feat,   // [N_NODES, 128]
    const float* __restrict__ edge_feat,   // [E, 128]
    const float* __restrict__ W,           // [384, 128]
    const float* __restrict__ bias,        // [128]
    const int*   __restrict__ src,         // [E]
    const int*   __restrict__ dst,         // [E]
    float*       __restrict__ out,         // [E, 128]
    int nEdges)
{
    // W in bf16, transposed to N-major: Wlds[n*384 + k]. 96 KB of LDS.
    extern __shared__ __bf16 Wlds[];

    const int tid = threadIdx.x;

    // ---- Stage W -> LDS (one-time per workgroup) --------------------------
    // Consecutive threads walk consecutive k => ~conflict-free b16 ds stores.
    // The strided global reads of W stay L2-resident (196 KB << 192 MB).
    for (int idx = tid; idx < KTOT * D_NODE; idx += TPB) {
        const int k = idx % KTOT;
        const int n = idx / KTOT;
        Wlds[n * KTOT + k] = (__bf16)W[k * D_NODE + n];
    }
    __syncthreads();

    const int lane = tid & 31;
    const int wave = tid >> 5;
    const int nn   = lane & 15;   // column / row-in-tile index
    const int hi   = lane >> 4;   // lane-half selector

    // Per-lane bias values for the 8 N-tiles (column = t*16 + nn).
    float bcol[8];
#pragma unroll
    for (int t = 0; t < 8; ++t) bcol[t] = bias[t * 16 + nn];

    const int nBlocks = (nEdges + 15) >> 4;

    // Persistent loop: each wave owns 16-edge x 128-col output tiles.
    for (int blk = blockIdx.x * WAVES + wave; blk < nBlocks;
         blk += gridDim.x * WAVES) {

        const int rowA = blk * 16 + nn;                 // this lane's edge row
        const int e    = rowA < nEdges ? rowA : nEdges - 1;  // clamp (EXEC all-1)

        // Three source regions of the virtual concat row.
        const float* __restrict__ base0 = node_feat + (size_t)src[e] * D_NODE;
        const float* __restrict__ base1 = node_feat + (size_t)dst[e] * D_NODE;
        const float* __restrict__ base2 = edge_feat + (size_t)e      * D_NODE;

        v8f acc[8];
#pragma unroll
        for (int t = 0; t < 8; ++t) acc[t] = (v8f)(0.0f);

#pragma unroll
        for (int r = 0; r < 3; ++r) {
            const float* __restrict__ bp = (r == 0) ? base0
                                         : (r == 1) ? base1 : base2;
#pragma unroll
            for (int ks = 0; ks < 4; ++ks) {
                const int kstep = r * 4 + ks;           // 0..11, K-chunk of 32

                // ---- A fragment (16-bit A layout, CDNA5 ISA):
                // lanes 0-15 : M=nn, K = {kb..kb+7} U {kb+16..kb+23}, kb=ks*32
                // lanes 16-31: M=nn, K = {kb+8..kb+15} U {kb+24..kb+31}
                const int kloc = ks * 32 + hi * 8;
                const float4 f0 = *reinterpret_cast<const float4*>(bp + kloc);
                const float4 f1 = *reinterpret_cast<const float4*>(bp + kloc + 4);
                const float4 f2 = *reinterpret_cast<const float4*>(bp + kloc + 16);
                const float4 f3 = *reinterpret_cast<const float4*>(bp + kloc + 20);
                v16bf a;
                a[0]  = (__bf16)f0.x; a[1]  = (__bf16)f0.y;
                a[2]  = (__bf16)f0.z; a[3]  = (__bf16)f0.w;
                a[4]  = (__bf16)f1.x; a[5]  = (__bf16)f1.y;
                a[6]  = (__bf16)f1.z; a[7]  = (__bf16)f1.w;
                a[8]  = (__bf16)f2.x; a[9]  = (__bf16)f2.y;
                a[10] = (__bf16)f2.z; a[11] = (__bf16)f2.w;
                a[12] = (__bf16)f3.x; a[13] = (__bf16)f3.y;
                a[14] = (__bf16)f3.z; a[15] = (__bf16)f3.w;

                // ---- B fragments from LDS (lanes 0-15: K 0-7,16-23; 16-31: K 8-15,24-31)
                // Launder the K offset so the ds_loads are NOT loop-invariant:
                // without this, LICM hoists all 96 B fragments out of the blk
                // loop and spills them to scratch (seen in round-1 asm).
                int kbB = kstep * 32 + hi * 8;    // K {kb+8h..+7} U {kb+16+8h..+7}
                asm volatile("" : "+v"(kbB));
#pragma unroll
                for (int t = 0; t < 8; ++t) {
                    const int n = t * 16 + nn;
                    const __bf16* wp = &Wlds[n * KTOT + kbB];
                    union { uint4 q[2]; v16bf v; } bu;
                    bu.q[0] = *reinterpret_cast<const uint4*>(wp);      // ds_load_b128
                    bu.q[1] = *reinterpret_cast<const uint4*>(wp + 16); // ds_load_b128
                    acc[t] = __builtin_amdgcn_wmma_f32_16x16x32_bf16(
                        /*neg_a=*/false, a, /*neg_b=*/false, bu.v,
                        /*c_mod=*/(short)0, acc[t],
                        /*reuse_a=*/false, /*reuse_b=*/false);
                }
            }
        }

        // ---- Epilogue: bias + ReLU + store.
        // C/D layout: lanes 0-15 -> M = v (VGPR idx), lanes 16-31 -> M = 8+v.
#pragma unroll
        for (int t = 0; t < 8; ++t) {
#pragma unroll
            for (int v = 0; v < 8; ++v) {
                const int row = blk * 16 + hi * 8 + v;
                if (row < nEdges) {
                    float val = acc[t][v] + bcol[t];
                    val = val > 0.0f ? val : 0.0f;
                    out[(size_t)row * D_NODE + t * 16 + nn] = val;
                }
            }
        }
    }
}

extern "C" void kernel_launch(void* const* d_in, const int* in_sizes, int n_in,
                              void* d_out, int out_size, void* d_ws, size_t ws_size,
                              hipStream_t stream) {
    const float* node_feat = (const float*)d_in[0];
    const float* edge_feat = (const float*)d_in[1];
    const float* W         = (const float*)d_in[2];
    const float* bias      = (const float*)d_in[3];
    const int*   src       = (const int*)d_in[4];
    const int*   dst       = (const int*)d_in[5];
    float*       out       = (float*)d_out;

    const int nEdges  = in_sizes[4];            // E = len(src)
    const int nBlocks = (nEdges + 15) / 16;

    int grid = (nBlocks + WAVES - 1) / WAVES;   // persistent workgroups
    if (grid > 512) grid = 512;

    const size_t ldsBytes = (size_t)KTOT * D_NODE * sizeof(__bf16);  // 96 KB

    agg_bond_wmma<<<dim3(grid), dim3(TPB), ldsBytes, stream>>>(
        node_feat, edge_feat, W, bias, src, dst, out, nEdges);
}
